// DeepseekMLP_16587163697454
// MI455X (gfx1250) — matrix operand order repaired and checked by reading, not compiled, not run on hardware
//
#include <hip/hip_runtime.h>

// ---------------------------------------------------------------------------
// MI455X (gfx1250) int4-quantized MLP: out = silu(x@Wg^T) * (x@Wu^T) @ Wd^T
// ~1.1 TFLOP of GEMM vs ~20us of HBM at 23.3 TB/s -> matrix-core bound.
// All three GEMMs run on v_wmma_f32_16x16x32_f16 (f32 accum). int4 weights
// are dequantized to fp16 into LDS with the 0x6400 exponent trick: one
// v_pk_fma_f16 per nibble-pair (bf16 can't do this; fp16 mantissa can hold
// 1024+n exactly). LDS is double-buffered; stage i+1 movement overlaps
// stage i WMMA:
//   - A tiles: CDNA5 async copy global_load_async_to_lds_b128 (ASYNCcnt),
//     no VGPR staging, synchronized with s_wait_asynccnt + barrier.
//   - Weight tiles: global loads issued before compute (loadcnt waits hide
//     under WMMA), pk-fma dequant + ds_store after compute into back buffer.
// ---------------------------------------------------------------------------

typedef unsigned short u16;
typedef unsigned int   u32;
typedef unsigned long long u64;
typedef __attribute__((ext_vector_type(16))) _Float16 v16h;
typedef __attribute__((ext_vector_type(2)))  _Float16 h2;
typedef __attribute__((ext_vector_type(8)))  float    v8f;

#define HIDDEN 4096
#define INTER  11008
#define MTOK   4096      // B*S = 4*1024

#define BM  128
#define BN  64
#define BK  64           // == quantization GROUP size
#define PAD 8
#define LDK (BK + PAD)   // 72 u16 per row -> 144B stride (16B aligned, skewed)

__device__ __forceinline__ u16 f2h(float f) {
    return __builtin_bit_cast(u16, (_Float16)f);       // v_cvt_f16_f32 (RTE)
}
__device__ __forceinline__ u32 h2dup(float f) {        // {h,h} packed
    u32 h = f2h(f);
    return h | (h << 16);
}

union FragH { v16h v; uint4 u[2]; };

// 4x 16B/lane async copies: LDS[la+off] = MEM[ga+off] (INST_OFFSET applies to
// both sides per the ISA), tracked on ASYNCcnt.
__device__ __forceinline__ void async_copy64B(u32 la, u64 ga) {
    asm volatile("global_load_async_to_lds_b128 %0, %1, off"           :: "v"(la), "v"(ga) : "memory");
    asm volatile("global_load_async_to_lds_b128 %0, %1, off offset:16" :: "v"(la), "v"(ga) : "memory");
    asm volatile("global_load_async_to_lds_b128 %0, %1, off offset:32" :: "v"(la), "v"(ga) : "memory");
    asm volatile("global_load_async_to_lds_b128 %0, %1, off offset:48" :: "v"(la), "v"(ga) : "memory");
}
__device__ __forceinline__ void wait_async0() {
    asm volatile("s_wait_asynccnt 0x0" ::: "memory");
}

// dequant one packed byte -> two fp16: w = (nib - z)*s via pk_fma
// low half = hi nibble (even k), high half = lo nibble (odd k)
__device__ __forceinline__ u32 dq2(u32 b, h2 s2, h2 c2) {
    u32 raw = 0x64006400u | ((b & 0xF0u) >> 4) | ((b & 0x0Fu) << 16);
    h2 w = __builtin_bit_cast(h2, raw) * s2 + c2;      // v_pk_fma_f16
    return __builtin_bit_cast(u32, w);
}

// --------------------------- x : fp32 -> fp16 ------------------------------
__global__ __launch_bounds__(256)
void dsmlp_xcast(const float* __restrict__ x, u16* __restrict__ y, int n) {
    int i = (blockIdx.x * blockDim.x + threadIdx.x) * 8;
    if (i >= n) return;
    const float4* p = reinterpret_cast<const float4*>(x + i);
    float4 a = p[0], b = p[1];
    uint4 o;
    o.x = (u32)f2h(a.x) | ((u32)f2h(a.y) << 16);
    o.y = (u32)f2h(a.z) | ((u32)f2h(a.w) << 16);
    o.z = (u32)f2h(b.x) | ((u32)f2h(b.y) << 16);
    o.w = (u32)f2h(b.z) | ((u32)f2h(b.w) << 16);
    *reinterpret_cast<uint4*>(y + i) = o;
}

// ----------------- fused gate/up GEMM + SiLU*mul epilogue ------------------
__global__ __launch_bounds__(256)
void dsmlp_gateup(const u16* __restrict__ X,
                  const int* __restrict__ gWq, const int* __restrict__ uWq,
                  const float* __restrict__ gS, const float* __restrict__ gZ,
                  const float* __restrict__ uS, const float* __restrict__ uZ,
                  u16* __restrict__ H)
{
    __shared__ u16 sA[2][BM][LDK];   // 36 KB
    __shared__ u16 sG[2][BN][LDK];   // 18 KB
    __shared__ u16 sU[2][BN][LDK];   // 18 KB

    const int tid     = threadIdx.x;
    const int lane    = tid & 31;
    const int wave    = tid >> 5;     // 8 waves
    const int wm      = wave >> 1;    // 0..3  (32 rows each)
    const int wn      = wave & 1;     // 0..1  (32 cols each)
    const int lane_lo = lane & 15;
    const int hi      = lane >> 4;

    const int n0 = blockIdx.x * BN;   // inter dim
    const int m0 = blockIdx.y * BM;   // token dim

    const int arow = tid >> 1;        // 128 rows, 2 thr/row, 64B each
    const int aseg = (tid & 1) * 32;
    const int wrow = tid >> 2;        // 64 rows, 4 thr/row, 8 packed bytes each
    const int wseg = (tid & 3) * 8;

    const int nrow = n0 + wrow;
    const int* gbase = gWq + (size_t)nrow * (HIDDEN / 2) + wseg;
    const int* ubase = uWq + (size_t)nrow * (HIDDEN / 2) + wseg;

    auto issueA = [&](int k0, int buf) {
        u32 la = (u32)(size_t)&sA[buf][arow][aseg];
        u64 ga = (u64)(X + (size_t)(m0 + arow) * HIDDEN + k0 + aseg);
        async_copy64B(la, ga);
    };
    auto issueW = [&](int k0, int4& g0, int4& g1, int4& u0, int4& u1,
                      float& gsc, float& gzp, float& usc, float& uzp) {
        const int grp = k0 >> 6;
        gsc = gS[nrow * (HIDDEN / 64) + grp];
        gzp = gZ[nrow * (HIDDEN / 64) + grp];
        usc = uS[nrow * (HIDDEN / 64) + grp];
        uzp = uZ[nrow * (HIDDEN / 64) + grp];
        const int4* gp = reinterpret_cast<const int4*>(gbase + (k0 >> 1));
        const int4* up = reinterpret_cast<const int4*>(ubase + (k0 >> 1));
        g0 = gp[0]; g1 = gp[1];
        u0 = up[0]; u1 = up[1];
    };
    auto storeW = [&](int buf, int4 g0, int4 g1, int4 u0, int4 u1,
                      float gsc, float gzp, float usc, float uzp) {
        h2 gs2 = __builtin_bit_cast(h2, h2dup(gsc));
        h2 gc2 = __builtin_bit_cast(h2, h2dup(-(1024.0f + gzp) * gsc));
        h2 us2 = __builtin_bit_cast(h2, h2dup(usc));
        h2 uc2 = __builtin_bit_cast(h2, h2dup(-(1024.0f + uzp) * usc));
        int gb[8] = {g0.x, g0.y, g0.z, g0.w, g1.x, g1.y, g1.z, g1.w};
        int ub[8] = {u0.x, u0.y, u0.z, u0.w, u1.x, u1.y, u1.z, u1.w};
        u32* gl = reinterpret_cast<u32*>(&sG[buf][wrow][2 * wseg]);
        u32* ul = reinterpret_cast<u32*>(&sU[buf][wrow][2 * wseg]);
#pragma unroll
        for (int i = 0; i < 8; ++i) {
            gl[i] = dq2((u32)gb[i] & 0xFFu, gs2, gc2);
            ul[i] = dq2((u32)ub[i] & 0xFFu, us2, uc2);
        }
    };

    v8f accg[2][2], accu[2][2];
#pragma unroll
    for (int i = 0; i < 2; ++i)
#pragma unroll
        for (int j = 0; j < 2; ++j) { accg[i][j] = (v8f)0.0f; accu[i][j] = (v8f)0.0f; }

    constexpr int KSTAGES = HIDDEN / BK;     // 64

    // ---- prologue: stage 0 into buffer 0 ----
    {
        int4 g0, g1, u0, u1; float gsc, gzp, usc, uzp;
        issueA(0, 0);
        issueW(0, g0, g1, u0, u1, gsc, gzp, usc, uzp);
        storeW(0, g0, g1, u0, u1, gsc, gzp, usc, uzp);
    }

    for (int it = 0; it < KSTAGES; ++it) {
        const int buf = it & 1;
        wait_async0();          // stage-it async copies (issued last iter) done
        __syncthreads();        // + everyone's ds_stores visible

        const bool has_next = (it + 1) < KSTAGES;
        int4 g0, g1, u0, u1; float gsc, gzp, usc, uzp;
        if (has_next) {
            issueA((it + 1) * BK, buf ^ 1);                  // overlaps WMMA
            issueW((it + 1) * BK, g0, g1, u0, u1, gsc, gzp, usc, uzp);
        }

        // ---- compute stage it: 2 WMMA K-steps of 32 ----
#pragma unroll
        for (int kk = 0; kk < BK; kk += 32) {
            FragH fa[2], fg[2], fu[2];
#pragma unroll
            for (int mi = 0; mi < 2; ++mi) {
                const int r = wm * 32 + mi * 16 + lane_lo;
                // A 16x32 f16 frag, contiguous k (as B): lanes0-15 K 0-15, lanes16-31 K 16-31
                fa[mi].u[0] = *reinterpret_cast<const uint4*>(&sA[buf][r][kk + hi * 16]);
                fa[mi].u[1] = *reinterpret_cast<const uint4*>(&sA[buf][r][kk + hi * 16 + 8]);
            }
#pragma unroll
            for (int ni = 0; ni < 2; ++ni) {
                const int c = wn * 32 + ni * 16 + lane_lo;
                // B 32x16: lane = N column; lanes0-15 K 0-15, lanes16-31 K 16-31
                const uint4* qg = reinterpret_cast<const uint4*>(&sG[buf][c][kk + hi * 16]);
                fg[ni].u[0] = qg[0]; fg[ni].u[1] = qg[1];
                const uint4* qu = reinterpret_cast<const uint4*>(&sU[buf][c][kk + hi * 16]);
                fu[ni].u[0] = qu[0]; fu[ni].u[1] = qu[1];
            }
#pragma unroll
            for (int mi = 0; mi < 2; ++mi)
#pragma unroll
                for (int ni = 0; ni < 2; ++ni) {
                    accg[mi][ni] = __builtin_amdgcn_wmma_f32_16x16x32_f16(
                        false, fa[mi].v, false, fg[ni].v, (short)0, accg[mi][ni], false, false);
                    accu[mi][ni] = __builtin_amdgcn_wmma_f32_16x16x32_f16(
                        false, fa[mi].v, false, fu[ni].v, (short)0, accu[mi][ni], false, false);
                }
        }

        if (has_next)
            storeW(buf ^ 1, g0, g1, u0, u1, gsc, gzp, usc, uzp);
    }

    // ---- epilogue: h = silu(g)*u -> fp16 ----
#pragma unroll
    for (int mi = 0; mi < 2; ++mi)
#pragma unroll
        for (int ni = 0; ni < 2; ++ni) {
            const int rbase = m0 + wm * 32 + mi * 16 + hi * 8;   // C/D layout
            const int col   = n0 + wn * 32 + ni * 16 + lane_lo;
#pragma unroll
            for (int r = 0; r < 8; ++r) {
                float g = accg[mi][ni][r];
                float u = accu[mi][ni][r];
                float h = (g / (1.0f + __expf(-g))) * u;
                H[(size_t)(rbase + r) * INTER + col] = f2h(h);
            }
        }
}

// --------------------------- down GEMM (fp32 out) --------------------------
__global__ __launch_bounds__(256)
void dsmlp_down(const u16* __restrict__ Hin,
                const int* __restrict__ dWq,
                const float* __restrict__ dS, const float* __restrict__ dZ,
                float* __restrict__ out)
{
    __shared__ u16 sA[2][BM][LDK];
    __shared__ u16 sB[2][BN][LDK];

    const int tid     = threadIdx.x;
    const int lane    = tid & 31;
    const int wave    = tid >> 5;
    const int wm      = wave >> 1;
    const int wn      = wave & 1;
    const int lane_lo = lane & 15;
    const int hi      = lane >> 4;

    const int n0 = blockIdx.x * BN;   // hidden dim
    const int m0 = blockIdx.y * BM;   // token dim

    const int arow = tid >> 1;
    const int aseg = (tid & 1) * 32;
    const int wrow = tid >> 2;
    const int wseg = (tid & 3) * 8;

    const int nrow = n0 + wrow;
    const int* wbase = dWq + (size_t)nrow * (INTER / 2) + wseg;

    auto issueA = [&](int k0, int buf) {
        u32 la = (u32)(size_t)&sA[buf][arow][aseg];
        u64 ga = (u64)(Hin + (size_t)(m0 + arow) * INTER + k0 + aseg);
        async_copy64B(la, ga);
    };
    auto issueW = [&](int k0, int4& w0, int4& w1, float& sc, float& zp) {
        const int grp = k0 >> 6;
        sc = dS[nrow * (INTER / 64) + grp];
        zp = dZ[nrow * (INTER / 64) + grp];
        const int4* wp = reinterpret_cast<const int4*>(wbase + (k0 >> 1));
        w0 = wp[0]; w1 = wp[1];
    };
    auto storeW = [&](int buf, int4 w0, int4 w1, float sc, float zp) {
        h2 s2 = __builtin_bit_cast(h2, h2dup(sc));
        h2 c2 = __builtin_bit_cast(h2, h2dup(-(1024.0f + zp) * sc));
        int wb[8] = {w0.x, w0.y, w0.z, w0.w, w1.x, w1.y, w1.z, w1.w};
        u32* bl = reinterpret_cast<u32*>(&sB[buf][wrow][2 * wseg]);
#pragma unroll
        for (int i = 0; i < 8; ++i)
            bl[i] = dq2((u32)wb[i] & 0xFFu, s2, c2);
    };

    v8f acc[2][2];
#pragma unroll
    for (int i = 0; i < 2; ++i)
#pragma unroll
        for (int j = 0; j < 2; ++j) acc[i][j] = (v8f)0.0f;

    constexpr int KSTAGES = INTER / BK;      // 172

    {
        int4 w0, w1; float sc, zp;
        issueA(0, 0);
        issueW(0, w0, w1, sc, zp);
        storeW(0, w0, w1, sc, zp);
    }

    for (int it = 0; it < KSTAGES; ++it) {
        const int buf = it & 1;
        wait_async0();
        __syncthreads();

        const bool has_next = (it + 1) < KSTAGES;
        int4 w0, w1; float sc, zp;
        if (has_next) {
            issueA((it + 1) * BK, buf ^ 1);
            issueW((it + 1) * BK, w0, w1, sc, zp);
        }

#pragma unroll
        for (int kk = 0; kk < BK; kk += 32) {
            FragH fa[2], fb[2];
#pragma unroll
            for (int mi = 0; mi < 2; ++mi) {
                const int r = wm * 32 + mi * 16 + lane_lo;
                fa[mi].u[0] = *reinterpret_cast<const uint4*>(&sA[buf][r][kk + hi * 16]);
                fa[mi].u[1] = *reinterpret_cast<const uint4*>(&sA[buf][r][kk + hi * 16 + 8]);
            }
#pragma unroll
            for (int ni = 0; ni < 2; ++ni) {
                const int c = wn * 32 + ni * 16 + lane_lo;
                const uint4* q = reinterpret_cast<const uint4*>(&sB[buf][c][kk + hi * 16]);
                fb[ni].u[0] = q[0]; fb[ni].u[1] = q[1];
            }
#pragma unroll
            for (int mi = 0; mi < 2; ++mi)
#pragma unroll
                for (int ni = 0; ni < 2; ++ni)
                    acc[mi][ni] = __builtin_amdgcn_wmma_f32_16x16x32_f16(
                        false, fa[mi].v, false, fb[ni].v, (short)0, acc[mi][ni], false, false);
        }

        if (has_next)
            storeW(buf ^ 1, w0, w1, sc, zp);
    }

#pragma unroll
    for (int mi = 0; mi < 2; ++mi)
#pragma unroll
        for (int ni = 0; ni < 2; ++ni) {
            const int rbase = m0 + wm * 32 + mi * 16 + hi * 8;
            const int col   = n0 + wn * 32 + ni * 16 + lane_lo;
#pragma unroll
            for (int r = 0; r < 8; ++r)
                out[(size_t)(rbase + r) * HIDDEN + col] = acc[mi][ni][r];
        }
}

// ---------------------------------------------------------------------------
extern "C" void kernel_launch(void* const* d_in, const int* in_sizes, int n_in,
                              void* d_out, int out_size, void* d_ws, size_t ws_size,
                              hipStream_t stream) {
    const float* x   = (const float*)d_in[0];
    const int*   gWq = (const int*)d_in[1];
    const int*   uWq = (const int*)d_in[2];
    const int*   dWq = (const int*)d_in[3];
    const float* gS  = (const float*)d_in[4];
    const float* gZ  = (const float*)d_in[5];
    const float* uS  = (const float*)d_in[6];
    const float* uZ  = (const float*)d_in[7];
    const float* dS  = (const float*)d_in[8];
    const float* dZ  = (const float*)d_in[9];
    float* out = (float*)d_out;

    // workspace: X16 (32 MB fp16) | H (86 MB fp16)
    u16* X16 = (u16*)d_ws;
    u16* H   = (u16*)((char*)d_ws + (size_t)MTOK * HIDDEN * sizeof(u16));

    const int nX = MTOK * HIDDEN;                 // 16.7M elements
    dsmlp_xcast<<<nX / (256 * 8), 256, 0, stream>>>(x, X16, nX);

    dim3 g1(INTER / BN, MTOK / BM);               // (172, 32)
    dsmlp_gateup<<<g1, 256, 0, stream>>>(X16, gWq, uWq, gS, gZ, uS, uZ, H);

    dim3 g2(HIDDEN / BN, MTOK / BM);              // (64, 32)
    dsmlp_down<<<g2, 256, 0, stream>>>(H, dWq, dS, dZ, out);
}
